// ProposedModel_14259291423429
// MI455X (gfx1250) — matrix operand order repaired and checked by reading, not compiled, not run on hardware
//
#include <hip/hip_runtime.h>
#include <hip/hip_bf16.h>

// bf16 / f32 vector types for WMMA fragments (wave32, gfx1250)
typedef __attribute__((ext_vector_type(16))) __bf16 v16bf;
typedef __attribute__((ext_vector_type(2)))  __bf16 bf16x2;
typedef __attribute__((ext_vector_type(8)))  float  v8f;

// ---------------------------------------------------------------------------
// Weight pre-pack: fp32 OIHW -> bf16, laid out in WMMA A-fragment order.
// Block (mt,kt) = 1KB: 32 lanes x 16 bf16, k order shared with the
// B-tile loads (contiguous k = 16*half + e, not AMD's table order):
//   lane 0-15 : M = mt*16+lane,    K = kt*32 + {0..15}
//   lane16-31 : M = mt*16+lane-16, K = kt*32 + {16..31}
// Out-of-range M/K padded with zeros so the conv kernel needs no edge checks.
// ---------------------------------------------------------------------------
__global__ void pack_weights_kernel(const float* __restrict__ w, __bf16* __restrict__ out,
                                    int Cout, int K, int Ktiles, long long total)
{
    long long idx = (long long)blockIdx.x * blockDim.x + threadIdx.x;
    if (idx >= total) return;
    int e    = (int)(idx & 15);
    int lane = (int)((idx >> 4) & 31);
    long long rest = idx >> 9;
    int kt = (int)(rest % Ktiles);
    int mt = (int)(rest / Ktiles);
    int m = mt * 16 + (lane & 15);
    int k = kt * 32 + (lane >> 4) * 16 + e;
    float v = (m < Cout && k < K) ? w[(size_t)m * K + k] : 0.0f;
    out[idx] = (__bf16)v;
}

// ---------------------------------------------------------------------------
// Implicit-GEMM conv, V_WMMA_F32_16X16X32_BF16.
//   GEMM: M = Cout, N = B*Ho*Wo, K = Cin*KH*KW    (K <= 1728 in this net)
// Workgroup = 128 threads (4 waves) -> 64(M) x 64(N) tile:
//   - full k->(ci,ky,kx) im2col table built once in LDS (8KB)
//   - 32x64 bf16 B tile double-buffered in LDS -> ONE barrier per k-step
//   - A fragments: contiguous 32B/lane loads from pre-packed bf16 weights,
//     reused across 4 N column-halves -> 4 WMMAs per wave per k-step
// Epilogue: +bias, optional ReLU, NCHW store into channel slice
// [yCoff,yCoff+Cout) of a yCtot-channel tensor (inception concat for free).
// ---------------------------------------------------------------------------
__global__ void __launch_bounds__(128)
conv_wmma_kernel(const float* __restrict__ x, const __bf16* __restrict__ wpack,
                 const float* __restrict__ bias, float* __restrict__ y,
                 int B, int Cin, int Hin, int Win,
                 int Cout, int Hout, int Wout,
                 int KH, int KW, int stride, int pad,
                 int Ktiles, int yCtot, int yCoff, int relu)
{
    __shared__ __align__(32) __bf16 Bs[2][32 * 64];        // [parity][n][k]
    __shared__ short kCi[2048];                            // k -> input channel
    __shared__ char  kKy[2048], kKx[2048];                 // k -> kernel y,x
    __shared__ int   nImg[64], nRem[64], nIy0[64], nIx0[64];

    const int t    = threadIdx.x;
    const int lane = t & 31;
    const int wave = t >> 5;
    const int lm   = lane & 15;
    const int half = lane >> 4;

    const int K    = Cin * KH * KW;
    const int KHW  = KH * KW;
    const int HWo  = Hout * Wout;
    const int Npix = B * HWo;

    const int nTile = blockIdx.y * 64;
    const int mt    = blockIdx.x * 4 + wave;       // this wave's 16-row M tile
    const int mTile = mt * 16;

    // ---- im2col k-decode table (whole K range, once per block)
    for (int kk = t; kk < K; kk += 128) {
        int ci = kk / KHW;
        int r2 = kk - ci * KHW;
        int ky = r2 / KW;
        kCi[kk] = (short)ci;
        kKy[kk] = (char)ky;
        kKx[kk] = (char)(r2 - ky * KW);
    }
    // ---- pixel decode for the 64 columns (once per block)
    if (t < 64) {
        int n = nTile + t;
        if (n < Npix) {
            int img = n / HWo;
            int rem = n - img * HWo;
            int oy  = rem / Wout;
            int ox  = rem - oy * Wout;
            nImg[t] = img; nRem[t] = rem;
            nIy0[t] = oy * stride - pad;
            nIx0[t] = ox * stride - pad;
        } else {
            nImg[t] = -1; nRem[t] = 0; nIy0[t] = 0; nIx0[t] = 0;
        }
    }
    __syncthreads();

    // packed A fragment pointer for this wave (v16bf = 32B = one lane fragment)
    const v16bf* __restrict__ afrag =
        (const v16bf*)wpack + (size_t)mt * Ktiles * 32 + lane;

    v8f acc0 = {}, acc1 = {}, acc2 = {}, acc3 = {};

    for (int kt = 0; kt < Ktiles; ++kt) {
        const int k0 = kt * 32;
        __bf16* bs = Bs[kt & 1];

        // ---- stage 32(K) x 64(N) bf16 B tile: 1024 dwords, 8 per thread
        #pragma unroll
        for (int i = 0; i < 8; ++i) {
            int slot = t + i * 128;          // 0..1023 == n*16 + kpair
            int n    = slot >> 4;
            int kp   = (slot & 15) << 1;
            float v0 = 0.0f, v1 = 0.0f;
            int img = nImg[n];
            if (img >= 0) {
                const float* __restrict__ xin = x + (size_t)img * Cin * Hin * Win;
                int iy0 = nIy0[n], ix0 = nIx0[n];
                #pragma unroll
                for (int u = 0; u < 2; ++u) {
                    int kk = k0 + kp + u;
                    if (kk < K) {
                        int iy = iy0 + (int)kKy[kk];
                        int ix = ix0 + (int)kKx[kk];
                        if (iy >= 0 && iy < Hin && ix >= 0 && ix < Win) {
                            float v = xin[((size_t)kCi[kk] * Hin + iy) * Win + ix];
                            if (u == 0) v0 = v; else v1 = v;
                        }
                    }
                }
            }
            bf16x2 pk; pk.x = (__bf16)v0; pk.y = (__bf16)v1;
            ((bf16x2*)bs)[slot] = pk;
        }
        __syncthreads();   // implies LDS fence: prior reads/writes complete

        // ---- fragments + 4 WMMAs per wave (A reused across 4 column-halves)
        if (kt + 1 < Ktiles)
            __builtin_prefetch(afrag + (size_t)(kt + 1) * 32, 0, 0);  // global_prefetch_b8
        v16bf a  = afrag[(size_t)kt * 32];
        v16bf b0 = *(const v16bf*)&bs[(lm)      * 32 + half * 16];
        v16bf b1 = *(const v16bf*)&bs[(16 + lm) * 32 + half * 16];
        v16bf b2 = *(const v16bf*)&bs[(32 + lm) * 32 + half * 16];
        v16bf b3 = *(const v16bf*)&bs[(48 + lm) * 32 + half * 16];
        acc0 = __builtin_amdgcn_wmma_f32_16x16x32_bf16(false, a, false, b0,
                                                       (short)0, acc0, false, false);
        acc1 = __builtin_amdgcn_wmma_f32_16x16x32_bf16(false, a, false, b1,
                                                       (short)0, acc1, false, false);
        acc2 = __builtin_amdgcn_wmma_f32_16x16x32_bf16(false, a, false, b2,
                                                       (short)0, acc2, false, false);
        acc3 = __builtin_amdgcn_wmma_f32_16x16x32_bf16(false, a, false, b3,
                                                       (short)0, acc3, false, false);
        // no trailing barrier: next iteration writes the other parity buffer
    }

    // ---- epilogue: bias (+ReLU), 16 rows x 4 column-halves
    const int mBase = mTile + half * 8;
    #pragma unroll
    for (int r = 0; r < 8; ++r) {
        int m = mBase + r;
        if (m < Cout) {
            float bv = bias[m];
            size_t chOff = (size_t)(yCoff + m) * HWo;
            #pragma unroll
            for (int cb = 0; cb < 4; ++cb) {
                int col = cb * 16 + lm;
                int img = nImg[col];
                if (img >= 0) {
                    float v = (cb == 0 ? acc0[r] : cb == 1 ? acc1[r]
                               : cb == 2 ? acc2[r] : acc3[r]) + bv;
                    if (relu) v = fmaxf(v, 0.0f);
                    y[(size_t)img * yCtot * HWo + chOff + nRem[col]] = v;
                }
            }
        }
    }
}

// ---------------------------------------------------------------------------
// Training-mode BatchNorm2d: pass 1, per-channel mean/var over (B,H,W)
// ---------------------------------------------------------------------------
__global__ void bn_reduce_kernel(const float* __restrict__ x, float* __restrict__ stats,
                                 int B, int C, int HW)
{
    const int c = blockIdx.x;
    float s = 0.0f, ss = 0.0f;
    for (int bb = 0; bb < B; ++bb) {
        const float* p = x + ((size_t)bb * C + c) * HW;
        for (int r = threadIdx.x; r < HW; r += 256) {
            float v = p[r];
            s += v; ss += v * v;
        }
    }
    __shared__ float sh0[256], sh1[256];
    sh0[threadIdx.x] = s; sh1[threadIdx.x] = ss;
    __syncthreads();
    for (int o = 128; o > 0; o >>= 1) {
        if ((int)threadIdx.x < o) {
            sh0[threadIdx.x] += sh0[threadIdx.x + o];
            sh1[threadIdx.x] += sh1[threadIdx.x + o];
        }
        __syncthreads();
    }
    if (threadIdx.x == 0) {
        float N = (float)(B * HW);
        float m = sh0[0] / N;
        stats[2 * c]     = m;
        stats[2 * c + 1] = sh1[0] / N - m * m;  // population variance
    }
}

// BatchNorm pass 2 (in place): x = x*scale + shift, one (b,c) plane per blockIdx.y
__global__ void bn_apply_kernel(float* __restrict__ x, const float* __restrict__ stats,
                                const float* __restrict__ g, const float* __restrict__ be,
                                int C, int HW)
{
    int plane = blockIdx.y;             // b*C + c
    int c = plane % C;
    float m  = stats[2 * c], v = stats[2 * c + 1];
    float sc = g[c] * rsqrtf(v + 1e-5f);
    float sh = be[c] - m * sc;
    float* p = x + (size_t)plane * HW;
    for (int i = blockIdx.x * 256 + threadIdx.x; i < HW; i += gridDim.x * 256)
        p[i] = p[i] * sc + sh;
}

// ---------------------------------------------------------------------------
// MaxPool2d (k,s,p); one (b,c) plane per blockIdx.y
// ---------------------------------------------------------------------------
__global__ void maxpool_kernel(const float* __restrict__ x, float* __restrict__ y,
                               int H, int W, int Kk, int S, int P, int Ho, int Wo)
{
    size_t plane = blockIdx.y;
    const float* p = x + plane * (size_t)H * W;
    float* q = y + plane * (size_t)Ho * Wo;
    int HWo = Ho * Wo;
    for (int i = blockIdx.x * 256 + threadIdx.x; i < HWo; i += gridDim.x * 256) {
        int ho = i / Wo, wo = i - ho * Wo;
        int y0 = ho * S - P, x0 = wo * S - P;
        float m = -3.402823466e38f;
        for (int ky = 0; ky < Kk; ++ky) {
            int iy = y0 + ky; if (iy < 0 || iy >= H) continue;
            for (int kx = 0; kx < Kk; ++kx) {
                int ix = x0 + kx; if (ix < 0 || ix >= W) continue;
                m = fmaxf(m, p[iy * W + ix]);
            }
        }
        q[i] = m;
    }
}

// MaxFeatureFusion: elementwise max of the 4 head feature maps
__global__ void max4_kernel(const float* __restrict__ a, const float* __restrict__ b,
                            const float* __restrict__ c, const float* __restrict__ d,
                            float* __restrict__ y, long long n)
{
    long long i = (long long)blockIdx.x * blockDim.x + threadIdx.x;
    if (i >= n) return;
    y[i] = fmaxf(fmaxf(a[i], b[i]), fmaxf(c[i], d[i]));
}

// AdaptiveAvgPool2d((1,1)) + flatten: mean over HW per (b,c)
__global__ void avgpool_kernel(const float* __restrict__ x, float* __restrict__ y,
                               int HW, int total)
{
    int i = blockIdx.x * blockDim.x + threadIdx.x;
    if (i >= total) return;
    const float* p = x + (size_t)i * HW;
    float s = 0.0f;
    for (int j = 0; j < HW; ++j) s += p[j];
    y[i] = s / (float)HW;
}

// Linear: y[b,o] = bias[o] + sum_k x[b,k] * W[o,k]   (W row-major [O,I])
__global__ void fc_kernel(const float* __restrict__ x, const float* __restrict__ w,
                          const float* __restrict__ b, float* __restrict__ y,
                          int Bn, int I, int O)
{
    int i = blockIdx.x * blockDim.x + threadIdx.x;
    if (i >= Bn * O) return;
    int bb = i / O, o = i - bb * O;
    const float* xr = x + (size_t)bb * I;
    const float* wr = w + (size_t)o * I;
    float s = b[o];
    for (int k = 0; k < I; ++k) s += xr[k] * wr[k];
    y[i] = s;
}

// ---------------------------------------------------------------------------
// Host orchestration. Param flattening order (setup_inputs dict order):
//   0..3: x1..x4
//   head h (h=0..3) base = 4 + 84*h:
//     +0 c1w +1 c1b +2 g1 +3 be1 +4 c2w +5 c2b +6 g2 +7 be2
//     +8 c3w +9 c3b +10 g3 +11 be3, then i1..i6 at +12,+24,+36,+48,+60,+72
//     (each inception: b1w b1b b2aw b2ab b2bw b2bb b3aw b3ab b3bw b3bb b4w b4b)
//   j1=340, j2=352, j3=364;  fc1w=376 fc1b=377 fc2w=378 fc2b=379
// ---------------------------------------------------------------------------
extern "C" void kernel_launch(void* const* d_in, const int* in_sizes, int n_in,
                              void* d_out, int out_size, void* d_ws, size_t ws_size,
                              hipStream_t stream)
{
    (void)in_sizes; (void)n_in; (void)out_size; (void)ws_size;

    auto P = [&](int i) { return (const float*)d_in[i]; };

    // ---- workspace bump allocator (floats, 32B-aligned chunks); peak ~132 MB
    float* wsf = (float*)d_ws;
    size_t off = 0;
    auto alloc = [&](size_t nelem) {
        float* p = wsf + off;
        off += (nelem + 7) & ~(size_t)7;   // keep 32B alignment for bf16 frags
        return p;
    };

    float* bufA  = alloc(16ull * 64 * 112 * 112);   // 12,845,056 f (conv1 out, ping)
    float* bufB  = alloc(16ull * 192 * 55 * 55);    //  9,292,800 f (pong)
    float* tmpP  = alloc(16ull * 256 * 27 * 27);    //  2,985,984 f (pool-branch temp)
    float* tmpR  = alloc(1600000);                  // reduce-branch temp (max 1,492,992)
    float* fh[4];
    for (int h = 0; h < 4; ++h) fh[h] = alloc(16ull * 528 * 13 * 13);  // 1,427,712 each
    __bf16* wpack = (__bf16*)alloc(524288);         // 1M bf16 = 2MB packed-weight scratch
    float* stats  = alloc(1024);                    // per-channel mean/var
    float* pooled = alloc(16 * 1024);
    float* fc1o   = alloc(16 * 512);

    // ---- helpers
    auto conv = [&](const float* xin, int wi, int bi, float* yout,
                    int Bn, int Cin, int Hin, int Win, int Cout,
                    int KH, int KW, int st, int pd,
                    int yCtot, int yCoff, int relu, int& Ho, int& Wo) {
        Ho = (Hin + 2 * pd - KH) / st + 1;
        Wo = (Win + 2 * pd - KW) / st + 1;
        int K      = Cin * KH * KW;                 // <= 1728 for every conv here
        int Ktiles = (K + 31) / 32;
        int MtP    = ((Cout + 63) / 64) * 4;        // M tiles incl. block padding
        long long ptot = (long long)MtP * Ktiles * 512;
        pack_weights_kernel<<<(unsigned)((ptot + 255) / 256), 256, 0, stream>>>(
            P(wi), wpack, Cout, K, Ktiles, ptot);
        int Npix = Bn * Ho * Wo;
        dim3 g((unsigned)((Cout + 63) / 64), (unsigned)((Npix + 63) / 64));
        conv_wmma_kernel<<<g, 128, 0, stream>>>(xin, wpack, P(bi), yout,
                                                Bn, Cin, Hin, Win, Cout, Ho, Wo,
                                                KH, KW, st, pd, Ktiles,
                                                yCtot, yCoff, relu);
    };

    auto bn = [&](float* xbuf, int gi, int bi, int Bn, int C, int HW) {
        bn_reduce_kernel<<<C, 256, 0, stream>>>(xbuf, stats, Bn, C, HW);
        dim3 g((unsigned)((HW + 255) / 256), (unsigned)(Bn * C));
        bn_apply_kernel<<<g, 256, 0, stream>>>(xbuf, stats, P(gi), P(bi), C, HW);
    };

    auto mpool = [&](const float* xin, float* yout, int Bn, int C, int H, int W,
                     int Kk, int S, int Pd, int& Ho, int& Wo) {
        Ho = (H + 2 * Pd - Kk) / S + 1;
        Wo = (W + 2 * Pd - Kk) / S + 1;
        dim3 g((unsigned)((Ho * Wo + 255) / 256), (unsigned)(Bn * C));
        maxpool_kernel<<<g, 256, 0, stream>>>(xin, yout, H, W, Kk, S, Pd, Ho, Wo);
    };

    auto incep = [&](const float* xin, int Bn, int Cin, int H, int W, int pb,
                     float* out, int c1, int c3r, int c3, int c5r, int c5, int pp) {
        int Ho, Wo;
        int Ctot = c1 + c3 + c5 + pp;
        conv(xin, pb + 0, pb + 1, out, Bn, Cin, H, W, c1, 1, 1, 1, 0, Ctot, 0, 1, Ho, Wo);
        conv(xin, pb + 2, pb + 3, tmpR, Bn, Cin, H, W, c3r, 1, 1, 1, 0, c3r, 0, 1, Ho, Wo);
        conv(tmpR, pb + 4, pb + 5, out, Bn, c3r, H, W, c3, 3, 3, 1, 1, Ctot, c1, 1, Ho, Wo);
        conv(xin, pb + 6, pb + 7, tmpR, Bn, Cin, H, W, c5r, 1, 1, 1, 0, c5r, 0, 1, Ho, Wo);
        conv(tmpR, pb + 8, pb + 9, out, Bn, c5r, H, W, c5, 5, 5, 1, 2, Ctot, c1 + c3, 1, Ho, Wo);
        mpool(xin, tmpP, Bn, Cin, H, W, 3, 1, 1, Ho, Wo);
        conv(tmpP, pb + 10, pb + 11, out, Bn, Cin, H, W, pp, 1, 1, 1, 0,
             Ctot, c1 + c3 + c5, 1, Ho, Wo);
    };

    auto head = [&](const float* xin, int pb, float* fout) {
        int Ho, Wo;
        conv(xin, pb + 0, pb + 1, bufA, 16, 3, 224, 224, 64, 7, 7, 2, 3, 64, 0, 0, Ho, Wo); // 112
        bn(bufA, pb + 2, pb + 3, 16, 64, Ho * Wo);
        mpool(bufA, bufB, 16, 64, 112, 112, 3, 2, 0, Ho, Wo);                               // 55
        conv(bufB, pb + 4, pb + 5, bufA, 16, 64, 55, 55, 64, 1, 1, 1, 0, 64, 0, 0, Ho, Wo);
        bn(bufA, pb + 6, pb + 7, 16, 64, 55 * 55);
        conv(bufA, pb + 8, pb + 9, bufB, 16, 64, 55, 55, 192, 3, 3, 1, 1, 192, 0, 0, Ho, Wo);
        bn(bufB, pb + 10, pb + 11, 16, 192, 55 * 55);
        mpool(bufB, bufA, 16, 192, 55, 55, 3, 2, 0, Ho, Wo);                                // 27
        incep(bufA, 16, 192, 27, 27, pb + 12, bufB, 64, 96, 128, 16, 32, 32);   // -> 256
        incep(bufB, 16, 256, 27, 27, pb + 24, bufA, 128, 128, 192, 32, 96, 64); // -> 480
        mpool(bufA, bufB, 16, 480, 27, 27, 3, 2, 0, Ho, Wo);                    // 13
        incep(bufB, 16, 480, 13, 13, pb + 36, bufA, 192, 96, 208, 16, 48, 64);  // -> 512
        incep(bufA, 16, 512, 13, 13, pb + 48, bufB, 160, 112, 224, 24, 64, 64); // -> 512
        incep(bufB, 16, 512, 13, 13, pb + 60, bufA, 128, 128, 256, 24, 64, 64); // -> 512
        incep(bufA, 16, 512, 13, 13, pb + 72, fout, 112, 144, 288, 32, 64, 64); // -> 528
    };

    // ---- 4 heads
    for (int h = 0; h < 4; ++h)
        head((const float*)d_in[h], 4 + 84 * h, fh[h]);

    // ---- max fusion -> [16,528,13,13]
    long long nf = 16ll * 528 * 13 * 13;
    max4_kernel<<<(unsigned)((nf + 255) / 256), 256, 0, stream>>>(
        fh[0], fh[1], fh[2], fh[3], bufA, nf);

    // ---- tail inceptions
    int Ho, Wo;
    incep(bufA, 16, 528, 13, 13, 340, bufB, 256, 160, 320, 32, 128, 128);  // -> 832 @13
    mpool(bufB, bufA, 16, 832, 13, 13, 3, 2, 0, Ho, Wo);                   // -> 6
    incep(bufA, 16, 832, 6, 6, 352, bufB, 256, 160, 320, 32, 128, 128);    // -> 832 @6
    incep(bufB, 16, 832, 6, 6, 364, bufA, 384, 192, 384, 48, 128, 128);    // -> 1024 @6

    // ---- avgpool + classifier
    avgpool_kernel<<<(16 * 1024 + 255) / 256, 256, 0, stream>>>(bufA, pooled, 36, 16 * 1024);
    fc_kernel<<<(16 * 512 + 255) / 256, 256, 0, stream>>>(pooled, P(376), P(377), fc1o, 16, 1024, 512);
    fc_kernel<<<(16 * 4 + 255) / 256, 256, 0, stream>>>(fc1o, P(378), P(379), (float*)d_out, 16, 512, 4);
}
